// BinaryQuadratic_25589415149863
// MI455X (gfx1250) — 8-bit matrix operand order repaired and checked by reading, compiled, not run on hardware
//
#include <hip/hip_runtime.h>

typedef __attribute__((ext_vector_type(8))) int   v8i;
typedef __attribute__((ext_vector_type(8))) float v8f;
typedef __attribute__((ext_vector_type(4))) float v4f;
typedef __attribute__((ext_vector_type(2))) float v2f;
typedef __attribute__((address_space(3)))  float lds_f;

#define BITS 4
#define RW   8
#define CW   8
#define YR   512
#define ID   64
#define ZC   512
#define NK   4096
#define NN   4096

#define BK   32
#define LDR  36
#define KCHUNKS (NK / BK)

__device__ __forceinline__ void async_cp16(lds_f* l, const float* g) {
  asm volatile("global_load_async_to_lds_b128 %0, %1, off"
               :: "v"(l), "v"(g) : "memory");
}

__global__ __launch_bounds__(128)
void bq_build_w(const float* __restrict__ Y, const float* __restrict__ Z,
                const float* __restrict__ Ac, const float* __restrict__ Bc,
                const float* __restrict__ Cc, const float* __restrict__ Dc,
                float* __restrict__ W) {
  const int lane = threadIdx.x & 31;
  const int wv   = threadIdx.x >> 5;
  const int tile = blockIdx.x * 4 + wv;
  const int ztg =  tile        & 3;
  const int yt  = (tile >> 2)  & 31;
  const int cw  = (tile >> 7)  & 7;
  const int rw  = (tile >> 10) & 7;

  const int m    = lane & 15;
  const int half = lane >> 4;

  const v8i ones = {0x01010101, 0x01010101, 0x01010101, 0x01010101,
                    0x01010101, 0x01010101, 0x01010101, 0x01010101};
  const v8i zero = {0, 0, 0, 0, 0, 0, 0, 0};

  v8i   av[BITS];
  float fa[BITS], fc[BITS];
  float yterm[8] = {0.f, 0.f, 0.f, 0.f, 0.f, 0.f, 0.f, 0.f};

#pragma unroll
  for (int bit = 0; bit < BITS; ++bit) {
    const int gidx = (bit * RW + rw) * CW + cw;
    const float* Yrow = Y + (size_t)gidx * (YR * ID) + (size_t)(yt * 16 + m) * ID;
#pragma unroll
    for (int v = 0; v < 8; ++v) {
      const int kb = ((v >> 1) << 4) + ((v & 1) << 2) + (half << 3);
      v4f f = *(const v4f*)(Yrow + kb);
      unsigned p = 0;
      p |= (f.x > 0.5f) ? 1u         : 0u;
      p |= (f.y > 0.5f) ? (1u << 8)  : 0u;
      p |= (f.z > 0.5f) ? (1u << 16) : 0u;
      p |= (f.w > 0.5f) ? (1u << 24) : 0u;
      av[bit][v] = (int)p;
    }
    fa[bit] = Ac[gidx];
    fc[bit] = Cc[gidx];
    const float fb = Bc[gidx];
    v8i ys = __builtin_amdgcn_wmma_i32_16x16x64_iu8(false, av[bit], false, ones,
                                                    zero, false, false);
#pragma unroll
    for (int j = 0; j < 8; ++j) yterm[j] += fb * (float)ys[j];
  }

  const float dv = Dc[rw * CW + cw];

  for (int zi = 0; zi < 8; ++zi) {
    const int zt = ztg * 8 + zi;
    float wf[8];
#pragma unroll
    for (int j = 0; j < 8; ++j) wf[j] = dv + yterm[j];

#pragma unroll
    for (int bit = 0; bit < BITS; ++bit) {
      const int gidx = (bit * RW + rw) * CW + cw;
      const float* Zcol = Z + (size_t)gidx * (ID * ZC) + (zt * 16 + m);
      v8i bv;
#pragma unroll
      for (int v = 0; v < 8; ++v) {
        const int kb = ((v >> 1) << 4) + ((v & 1) << 2) + (half << 3);
        unsigned p = 0;
#pragma unroll
        for (int bb = 0; bb < 4; ++bb) {
          float f = Zcol[(size_t)(kb + bb) * ZC];
          p |= (f > 0.5f) ? (1u << (8 * bb)) : 0u;
        }
        bv[v] = (int)p;
      }
      v8i core = __builtin_amdgcn_wmma_i32_16x16x64_iu8(false, av[bit], false, bv,
                                                        zero, false, false);
      v8i zs   = __builtin_amdgcn_wmma_i32_16x16x64_iu8(false, ones,    false, bv,
                                                        zero, false, false);
#pragma unroll
      for (int j = 0; j < 8; ++j)
        wf[j] += fa[bit] * (float)core[j] + fc[bit] * (float)zs[j];
    }

    const int row0 = rw * YR + yt * 16 + half * 8;
    const int col  = cw * ZC + zt * 16 + m;
#pragma unroll
    for (int j = 0; j < 8; ++j)
      W[(size_t)(row0 + j) * NK + col] = wf[j];
  }
}

__global__ __launch_bounds__(256)
void bq_gemm(const float* __restrict__ X, const float* __restrict__ W,
             const float* __restrict__ bias, float* __restrict__ out) {
  __shared__ __align__(16) float sX[2][128 * LDR];
  __shared__ __align__(16) float sW[2][128 * LDR];

  const int tid  = threadIdx.x;
  const int lane = tid & 31;
  const int wv   = tid >> 5;
  const int col  = lane & 15;
  const int half = lane >> 4;

  const int m0 = blockIdx.y * 128;
  const int n0 = blockIdx.x * 128;
  const int wm = (wv & 3) * 32;
  const int wn = (wv >> 2) * 64;

  v8f acc[2][4];
#pragma unroll
  for (int t = 0; t < 4; ++t) {
    const float bvi = bias[n0 + wn + t * 16 + col];
#pragma unroll
    for (int j = 0; j < 8; ++j) { acc[0][t][j] = bvi; acc[1][t][j] = bvi; }
  }

  auto stage = [&](int kc, int buf) {
    const int k0 = kc * BK;
#pragma unroll
    for (int i = 0; i < 4; ++i) {
      const int t = tid + i * 256;
      const int r = t >> 3;
      const int c = t & 7;
      async_cp16((lds_f*)&sX[buf][r * LDR + c * 4],
                 X + (size_t)(m0 + r) * NK + k0 + c * 4);
      async_cp16((lds_f*)&sW[buf][r * LDR + c * 4],
                 W + (size_t)(n0 + r) * NK + k0 + c * 4);
    }
  };

  stage(0, 0);

  for (int kc = 0; kc < KCHUNKS; ++kc) {
    const int cur = kc & 1;
    if (kc + 1 < KCHUNKS) {
      stage(kc + 1, cur ^ 1);
      asm volatile("s_wait_asynccnt 0x8" ::: "memory");
    } else {
      asm volatile("s_wait_asynccnt 0x0" ::: "memory");
    }
    __syncthreads();

#pragma unroll
    for (int kk = 0; kk < BK; kk += 4) {
      v2f a0 = *(const v2f*)&sX[cur][(wm      + col) * LDR + kk + half * 2];
      v2f a1 = *(const v2f*)&sX[cur][(wm + 16 + col) * LDR + kk + half * 2];
      v2f b0 = *(const v2f*)&sW[cur][(wn      + col) * LDR + kk + half * 2];
      v2f b1 = *(const v2f*)&sW[cur][(wn + 16 + col) * LDR + kk + half * 2];
      v2f b2 = *(const v2f*)&sW[cur][(wn + 32 + col) * LDR + kk + half * 2];
      v2f b3 = *(const v2f*)&sW[cur][(wn + 48 + col) * LDR + kk + half * 2];
      acc[0][0] = __builtin_amdgcn_wmma_f32_16x16x4_f32(false, a0, false, b0, (short)0, acc[0][0], false, false);
      acc[0][1] = __builtin_amdgcn_wmma_f32_16x16x4_f32(false, a0, false, b1, (short)0, acc[0][1], false, false);
      acc[0][2] = __builtin_amdgcn_wmma_f32_16x16x4_f32(false, a0, false, b2, (short)0, acc[0][2], false, false);
      acc[0][3] = __builtin_amdgcn_wmma_f32_16x16x4_f32(false, a0, false, b3, (short)0, acc[0][3], false, false);
      acc[1][0] = __builtin_amdgcn_wmma_f32_16x16x4_f32(false, a1, false, b0, (short)0, acc[1][0], false, false);
      acc[1][1] = __builtin_amdgcn_wmma_f32_16x16x4_f32(false, a1, false, b1, (short)0, acc[1][1], false, false);
      acc[1][2] = __builtin_amdgcn_wmma_f32_16x16x4_f32(false, a1, false, b2, (short)0, acc[1][2], false, false);
      acc[1][3] = __builtin_amdgcn_wmma_f32_16x16x4_f32(false, a1, false, b3, (short)0, acc[1][3], false, false);
    }
    __syncthreads();
  }

#pragma unroll
  for (int mi = 0; mi < 2; ++mi)
#pragma unroll
    for (int t = 0; t < 4; ++t)
#pragma unroll
      for (int j = 0; j < 8; ++j)
        out[(size_t)(m0 + wm + mi * 16 + half * 8 + j) * NN +
            (n0 + wn + t * 16 + col)] = acc[mi][t][j];
}

extern "C" void kernel_launch(void* const* d_in, const int* in_sizes, int n_in,
                              void* d_out, int out_size, void* d_ws, size_t ws_size,
                              hipStream_t stream) {
  const float* X    = (const float*)d_in[0];
  const float* Y    = (const float*)d_in[1];
  const float* Z    = (const float*)d_in[2];
  const float* a    = (const float*)d_in[3];
  const float* b    = (const float*)d_in[4];
  const float* c    = (const float*)d_in[5];
  const float* d    = (const float*)d_in[6];
  const float* bias = (const float*)d_in[7];

  float* W   = (float*)d_ws;
  float* out = (float*)d_out;

  bq_build_w<<<2048, 128, 0, stream>>>(Y, Z, a, b, c, d, W);
  bq_gemm<<<dim3(32, 32), 256, 0, stream>>>(X, W, bias, out);
}
